// FrameNetComplex01_44117904065146
// MI455X (gfx1250) — matrix operand order repaired and checked by reading, not compiled, not run on hardware
//
#include <hip/hip_runtime.h>
#include <hip/hip_bf16.h>

// ---------------------------------------------------------------------------
// FrameNet-style GNN layer on MI455X (gfx1250).
// Edge MLP (two 128x128 GEMMs) runs on v_wmma_f32_16x16x32_bf16 (wave32).
// ---------------------------------------------------------------------------

constexpr int kD   = 128;   // feature dim
constexpr int kDQ  = 32;    // Dq = D/4
constexpr int kTM  = 16;    // edges per tile (WMMA M)
constexpr int kLDH = kD + 8;    // padded bf16 LDS row (halves)  -> 272B rows, 16B aligned
constexpr int kLDF = kD + 4;    // padded f32 LDS row

typedef __attribute__((ext_vector_type(16))) __bf16 v16bf;
typedef __attribute__((ext_vector_type(8)))  __bf16 v8bf;
typedef __attribute__((ext_vector_type(8)))  float  v8f;

__device__ __forceinline__ unsigned short f2bf(float f) {
    unsigned int u = __float_as_uint(f);
    unsigned int r = u + 0x7FFFu + ((u >> 16) & 1u);   // round-to-nearest-even
    return (unsigned short)(r >> 16);
}
__device__ __forceinline__ float silu_f(float x) {
    return x / (1.0f + __expf(-x));
}
__device__ __forceinline__ void norm3(float& x, float& y, float& z) {
    float n = sqrtf(x * x + y * y + z * z);
    float inv = 1.0f / fmaxf(n, 1e-8f);
    x *= inv; y *= inv; z *= inv;
}

// ---------------------------------------------------------------------------
// K0: residue frames. residue_frame[r] = frame[clamp(r-1, 0, Nr-3)]
// frame rows: diff, cross, vertical (matches jnp.stack(..., axis=1)).
// ---------------------------------------------------------------------------
__global__ void frames_kernel(const float* __restrict__ pr,
                              float* __restrict__ frames, int Nr) {
    int r = blockIdx.x * blockDim.x + threadIdx.x;
    if (r >= Nr) return;
    int k = r - 1;
    k = max(0, min(k, Nr - 3));
    // end = u[k] = normalize(pos[k+1]-pos[k]); start = u[k+1]
    float e0 = pr[(k + 1) * 3 + 0] - pr[k * 3 + 0];
    float e1 = pr[(k + 1) * 3 + 1] - pr[k * 3 + 1];
    float e2 = pr[(k + 1) * 3 + 2] - pr[k * 3 + 2];
    norm3(e0, e1, e2);
    float s0 = pr[(k + 2) * 3 + 0] - pr[(k + 1) * 3 + 0];
    float s1 = pr[(k + 2) * 3 + 1] - pr[(k + 1) * 3 + 1];
    float s2 = pr[(k + 2) * 3 + 2] - pr[(k + 1) * 3 + 2];
    norm3(s0, s1, s2);
    float d0 = e0 - s0, d1 = e1 - s1, d2 = e2 - s2;
    norm3(d0, d1, d2);
    float c0 = e1 * s2 - e2 * s1;
    float c1 = e2 * s0 - e0 * s2;
    float c2 = e0 * s1 - e1 * s0;
    norm3(c0, c1, c2);
    float v0 = d1 * c2 - d2 * c1;
    float v1 = d2 * c0 - d0 * c2;
    float v2 = d0 * c1 - d1 * c0;
    norm3(v0, v1, v2);
    float* F = frames + (size_t)r * 9;
    F[0] = d0; F[1] = d1; F[2] = d2;
    F[3] = c0; F[4] = c1; F[5] = c2;
    F[6] = v0; F[7] = v1; F[8] = v2;
}

// ---------------------------------------------------------------------------
// K1: per-edge geometry: edge_diff and proj = edge_diff^T * frame
// ---------------------------------------------------------------------------
__global__ void edgeprep_kernel(const float* __restrict__ pl,
                                const float* __restrict__ pr,
                                const int* __restrict__ eli,
                                const int* __restrict__ eri,
                                const float* __restrict__ frames,
                                float* __restrict__ ediff,
                                float* __restrict__ proj, int E) {
    int e = blockIdx.x * blockDim.x + threadIdx.x;
    if (e >= E) return;
    int li = eli[e], ri = eri[e];
    float d0 = pl[li * 3 + 0] - pr[ri * 3 + 0];
    float d1 = pl[li * 3 + 1] - pr[ri * 3 + 1];
    float d2 = pl[li * 3 + 2] - pr[ri * 3 + 2];
    norm3(d0, d1, d2);
    const float* F = frames + (size_t)ri * 9;
    proj[e * 3 + 0] = d0 * F[0] + d1 * F[3] + d2 * F[6];
    proj[e * 3 + 1] = d0 * F[1] + d1 * F[4] + d2 * F[7];
    proj[e * 3 + 2] = d0 * F[2] + d1 * F[5] + d2 * F[8];
    ediff[e * 3 + 0] = d0;
    ediff[e * 3 + 1] = d1;
    ediff[e * 3 + 2] = d2;
}

// ---------------------------------------------------------------------------
// K2: pack Win1/Win2 (fp32, [L][128][128]) into bf16 WMMA B-operand layout:
// dest[((l*2+mat)*4 + kt)*8 + nt][lane][half]; lanes 0-15 hold K 0..7,16..23 of
// their column, lanes 16-31 hold K 8..15,24..31 (CDNA5 16-bit B layout).
// ---------------------------------------------------------------------------
__global__ void pack_kernel(const float* __restrict__ Win1,
                            const float* __restrict__ Win2,
                            unsigned short* __restrict__ packed) {
    int idx = blockIdx.x * blockDim.x + threadIdx.x;
    if (idx >= 2 * 2 * kD * kD) return;
    int n   = idx & 127;
    int k   = (idx >> 7) & 127;
    int mat = (idx >> 14) & 1;
    int l   = idx >> 15;
    const float* W = mat ? Win2 : Win1;
    float v = W[((size_t)l * kD + k) * kD + n];
    int kt = k >> 5, kin = k & 31;
    int nt = n >> 4, nin = n & 15;
    int lane = ((kin >> 3) & 1) ? (nin + 16) : nin;
    int half = (kin & 7) + ((kin >> 4) << 3);
    size_t dst = ((((size_t)(l * 2 + mat) * 4 + kt) * 8 + nt) * 32 + lane) * 16 + half;
    packed[dst] = f2bf(v);
}

__global__ void zero2_kernel(float* __restrict__ a, float* __restrict__ b, int n) {
    int i = blockIdx.x * blockDim.x + threadIdx.x;
    if (i < n) { a[i] = 0.0f; b[i] = 0.0f; }
}
__global__ void copy_kernel(float* __restrict__ dst, const float* __restrict__ src, int n) {
    int i = blockIdx.x * blockDim.x + threadIdx.x;
    if (i < n) dst[i] = src[i];
}

// ---------------------------------------------------------------------------
// 16x16x128 tile GEMM: A from LDS bf16 (row-major, padded rows), B from
// pre-swizzled global weights. 4 chained v_wmma_f32_16x16x32_bf16.
// ---------------------------------------------------------------------------
__device__ __forceinline__ v8f gemm_tile(const unsigned short* shA,
                                         const unsigned short* __restrict__ wB,
                                         int lane, int nt) {
    v8f c = {0.f, 0.f, 0.f, 0.f, 0.f, 0.f, 0.f, 0.f};
    int row = lane & 15;
    int koff = (lane >> 4) << 3;          // lanes 16-31 take K base +8 / +24
#pragma unroll
    for (int kt = 0; kt < 4; ++kt) {
        int kb = kt * 32 + koff;
        union { v16bf v; v8bf h[2]; } a;
        a.h[0] = *reinterpret_cast<const v8bf*>(shA + row * kLDH + kb);        // K kb..kb+7
        a.h[1] = *reinterpret_cast<const v8bf*>(shA + row * kLDH + kb + 16);   // K kb+16..kb+23
        v16bf b = *reinterpret_cast<const v16bf*>(wB + (((size_t)kt * 8 + nt) * 32 + lane) * 16);
        c = __builtin_amdgcn_wmma_f32_16x16x32_bf16(false, a.v, false, b,
                                                    (short)0, c, false, false);
    }
    return c;
}

// ---------------------------------------------------------------------------
// K3: one message-passing layer over a 16-edge tile.
// ---------------------------------------------------------------------------
__global__ __launch_bounds__(128)
void layer_kernel(const float* __restrict__ lrep, const float* __restrict__ rrep,
                  const int* __restrict__ eli, const int* __restrict__ eri,
                  const float* __restrict__ ediff, const float* __restrict__ proj,
                  const unsigned short* __restrict__ packedW,   // this layer's block
                  const float* __restrict__ bin1, const float* __restrict__ bin2,
                  const float* __restrict__ Ws1a, const float* __restrict__ bs1a,
                  const float* __restrict__ Ws1b, const float* __restrict__ bs1b,
                  const float* __restrict__ Ws2a, const float* __restrict__ bs2a,
                  const float* __restrict__ Ws2b, const float* __restrict__ bs2b,
                  const float* __restrict__ lv_in, float* __restrict__ lv_out,
                  int E) {
    __shared__ __align__(16) unsigned short sh_hsum[kTM * kLDH];
    __shared__ __align__(16) unsigned short sh_h1[kTM * kLDH];
    __shared__ float sh_h[kTM * kLDF];
    __shared__ float sh_proj[kTM * 3];
    __shared__ float sh_ed[kTM * 3];
    __shared__ float sh_s1[kTM * kDQ];
    __shared__ float sh_s2[kTM * kDQ];
    __shared__ float sh_bs1a[kDQ], sh_bs2a[kDQ], sh_w1b[kDQ], sh_w2b[kDQ];
    __shared__ int   sh_el[kTM];

    const int t = threadIdx.x;          // blockDim == 128 == kD
    const int tile = blockIdx.x;
    const int e0 = tile * kTM;
    int nvalid = E - e0;
    if (nvalid > kTM) nvalid = kTM;

    // ---- phase 1: gather h_sum (bf16), per-edge geometry, s-vectors -------
    for (int e = 0; e < kTM; ++e) {
        int ge = e0 + e;
        if (ge >= E) ge = E - 1;                 // clamp; masked at scatter
        int li = eli[ge], ri = eri[ge];
        float v = lrep[(size_t)li * kD + t] + rrep[(size_t)ri * kD + t];
        sh_hsum[e * kLDH + t] = f2bf(v);
        if (t == 0) sh_el[e] = li;
        if (t < 3) {
            sh_ed[e * 3 + t]   = ediff[(size_t)ge * 3 + t];
            sh_proj[e * 3 + t] = proj[(size_t)ge * 3 + t];
        }
    }
    if (t < kDQ) {
        sh_bs1a[t] = bs1a[t]; sh_bs2a[t] = bs2a[t];
        sh_w1b[t]  = Ws1b[t]; sh_w2b[t]  = Ws2b[t];
    }
    // s{1,2}[e][q] = sum_i proj[e][i] * Ws{1,2}a[i][q]
    for (int idx = t; idx < kTM * kDQ; idx += 128) {
        int e = idx / kDQ, q = idx % kDQ;
        int ge = e0 + e;
        if (ge >= E) ge = E - 1;
        float p0 = proj[(size_t)ge * 3 + 0];
        float p1 = proj[(size_t)ge * 3 + 1];
        float p2 = proj[(size_t)ge * 3 + 2];
        sh_s1[idx] = p0 * Ws1a[0 * kDQ + q] + p1 * Ws1a[1 * kDQ + q] + p2 * Ws1a[2 * kDQ + q];
        sh_s2[idx] = p0 * Ws2a[0 * kDQ + q] + p1 * Ws2a[1 * kDQ + q] + p2 * Ws2a[2 * kDQ + q];
    }
    __syncthreads();

    const int lane = t & 31;
    const int wave = t >> 5;
    const int colin = lane & 15;
    const int rowbase = (lane >> 4) << 3;

    // ---- phase 2a: h1 = silu(h_sum @ Win1 + bin1), bf16 into LDS ----------
#pragma unroll
    for (int w = 0; w < 2; ++w) {
        int nt = wave + w * 4;
        v8f c = gemm_tile(sh_hsum, packedW, lane, nt);      // mat 0
        int col = nt * 16 + colin;
        float bias = bin1[col];
#pragma unroll
        for (int j = 0; j < 8; ++j)
            sh_h1[(rowbase + j) * kLDH + col] = f2bf(silu_f(c[j] + bias));
    }
    __syncthreads();

    // ---- phase 2b: h = h1 @ Win2 + bin2, fp32 into LDS --------------------
    const unsigned short* wB2 = packedW + (size_t)4 * 8 * 32 * 16;   // mat 1
#pragma unroll
    for (int w = 0; w < 2; ++w) {
        int nt = wave + w * 4;
        v8f c = gemm_tile(sh_h1, wB2, lane, nt);
        int col = nt * 16 + colin;
        float bias = bin2[col];
#pragma unroll
        for (int j = 0; j < 8; ++j)
            sh_h[(rowbase + j) * kLDF + col] = c[j] + bias;
    }
    __syncthreads();

    // ---- phase 3: scalar-weight MLP + run-compressed atomic scatter -------
    // thread t owns feature d = t; walk edges (el is sorted => runs per node)
    const float b1b = bs1b[0], b2b = bs2b[0];
    const int d = t;
    int   cur = -1;
    float acc0 = 0.f, acc1 = 0.f, acc2 = 0.f;
    float li0 = 0.f, li1 = 0.f, li2 = 0.f;
    for (int e = 0; e < nvalid; ++e) {
        int node = sh_el[e];
        if (node != cur) {
            if (cur >= 0) {
                atomicAdd(&lv_out[(size_t)cur * 384 + 0 * kD + d], acc0);
                atomicAdd(&lv_out[(size_t)cur * 384 + 1 * kD + d], acc1);
                atomicAdd(&lv_out[(size_t)cur * 384 + 2 * kD + d], acc2);
            }
            cur = node; acc0 = acc1 = acc2 = 0.f;
            li0 = lv_in[(size_t)node * 384 + 0 * kD + d];
            li1 = lv_in[(size_t)node * 384 + 1 * kD + d];
            li2 = lv_in[(size_t)node * 384 + 2 * kD + d];
        }
        float h  = sh_h[e * kLDF + d];
        float w01 = b1b, w02 = b2b;
#pragma unroll
        for (int q = 0; q < kDQ; ++q) {
            float z1 = fmaf(h, sh_s1[e * kDQ + q], sh_bs1a[q]);
            float z2 = fmaf(h, sh_s2[e * kDQ + q], sh_bs2a[q]);
            w01 = fmaf(silu_f(z1), sh_w1b[q], w01);
            w02 = fmaf(silu_f(z2), sh_w2b[q], w02);
        }
        float hp = h * sh_proj[e * 3 + 0];      // perm[...,0] term
        w01 += hp; w02 += hp;
        acc0 = fmaf(li0, w01, fmaf(sh_ed[e * 3 + 0], w02, acc0));
        acc1 = fmaf(li1, w01, fmaf(sh_ed[e * 3 + 1], w02, acc1));
        acc2 = fmaf(li2, w01, fmaf(sh_ed[e * 3 + 2], w02, acc2));
    }
    if (cur >= 0) {
        atomicAdd(&lv_out[(size_t)cur * 384 + 0 * kD + d], acc0);
        atomicAdd(&lv_out[(size_t)cur * 384 + 1 * kD + d], acc1);
        atomicAdd(&lv_out[(size_t)cur * 384 + 2 * kD + d], acc2);
    }
}

// ---------------------------------------------------------------------------
// K4: out[n,i] = sum_d relu(lv[n,i,:]·Wl1[:,d] + bl1[d])·Wl2[d] + bl2 + lvi
// 4 (n,i)-rows per block to reuse Wl1 reads.
// ---------------------------------------------------------------------------
__global__ __launch_bounds__(128)
void head_kernel(const float* __restrict__ lv, const float* __restrict__ Wl1,
                 const float* __restrict__ bl1, const float* __restrict__ Wl2,
                 const float* __restrict__ bl2, const float* __restrict__ lvi,
                 float* __restrict__ out, int total) {
    __shared__ float row[4][kD];
    __shared__ float red[4][kD];
    int t = threadIdx.x;
    int base = blockIdx.x * 4;
#pragma unroll
    for (int r = 0; r < 4; ++r) {
        int nb = base + r;
        row[r][t] = (nb < total) ? lv[(size_t)nb * kD + t] : 0.0f;
    }
    __syncthreads();
    float acc[4] = {0.f, 0.f, 0.f, 0.f};
    for (int dp = 0; dp < kD; ++dp) {
        float w = Wl1[(size_t)dp * kD + t];
#pragma unroll
        for (int r = 0; r < 4; ++r) acc[r] = fmaf(row[r][dp], w, acc[r]);
    }
    float b = bl1[t], w2 = Wl2[t];
#pragma unroll
    for (int r = 0; r < 4; ++r) red[r][t] = fmaxf(acc[r] + b, 0.0f) * w2;
    __syncthreads();
    for (int s = 64; s > 0; s >>= 1) {
        if (t < s) {
#pragma unroll
            for (int r = 0; r < 4; ++r) red[r][t] += red[r][t + s];
        }
        __syncthreads();
    }
    if (t < 4) {
        int nb = base + t;
        if (nb < total) out[nb] = red[t][0] + bl2[0] + lvi[nb];
    }
}

// ---------------------------------------------------------------------------
extern "C" void kernel_launch(void* const* d_in, const int* in_sizes, int n_in,
                              void* d_out, int out_size, void* d_ws, size_t ws_size,
                              hipStream_t stream) {
    const float* lrep = (const float*)d_in[0];
    const float* lvi  = (const float*)d_in[1];
    const float* pl   = (const float*)d_in[2];
    const float* rrep = (const float*)d_in[3];
    const float* pr   = (const float*)d_in[4];
    const int*   eli  = (const int*)d_in[7];
    const int*   eri  = (const int*)d_in[8];
    const float* Win1 = (const float*)d_in[9];
    const float* bin1 = (const float*)d_in[10];
    const float* Win2 = (const float*)d_in[11];
    const float* bin2 = (const float*)d_in[12];
    const float* Ws1a = (const float*)d_in[13];
    const float* bs1a = (const float*)d_in[14];
    const float* Ws1b = (const float*)d_in[15];
    const float* bs1b = (const float*)d_in[16];
    const float* Ws2a = (const float*)d_in[17];
    const float* bs2a = (const float*)d_in[18];
    const float* Ws2b = (const float*)d_in[19];
    const float* bs2b = (const float*)d_in[20];
    const float* Wl1  = (const float*)d_in[21];
    const float* bl1  = (const float*)d_in[22];
    const float* Wl2  = (const float*)d_in[23];
    const float* bl2  = (const float*)d_in[24];

    const int Nl = in_sizes[2] / 3;
    const int Nr = in_sizes[4] / 3;
    const int E  = in_sizes[7];

    char* ws = (char*)d_ws;
    size_t off = 0;
    auto take = [&](size_t bytes) -> char* {
        char* p = ws + off;
        off = (off + bytes + 255) & ~(size_t)255;
        return p;
    };
    float* frames = (float*)take((size_t)Nr * 9 * 4);
    float* ediff  = (float*)take((size_t)E * 3 * 4);
    float* proj   = (float*)take((size_t)E * 3 * 4);
    unsigned short* packed = (unsigned short*)take((size_t)2 * 2 * 4 * 8 * 32 * 16 * 2);
    const int nlv = Nl * 384;
    float* lv0 = (float*)take((size_t)nlv * 4);
    float* lv1 = (float*)take((size_t)nlv * 4);
    float* lv2 = (float*)take((size_t)nlv * 4);

    zero2_kernel<<<(nlv + 255) / 256, 256, 0, stream>>>(lv0, lv1, nlv);
    frames_kernel<<<(Nr + 255) / 256, 256, 0, stream>>>(pr, frames, Nr);
    edgeprep_kernel<<<(E + 255) / 256, 256, 0, stream>>>(pl, pr, eli, eri, frames, ediff, proj, E);
    pack_kernel<<<(2 * 2 * kD * kD + 255) / 256, 256, 0, stream>>>(Win1, Win2, packed);

    const int ntiles = (E + kTM - 1) / kTM;
    for (int l = 0; l < 2; ++l) {
        const float* lvin = l ? lv1 : lv0;
        float* lvout = l ? lv2 : lv1;
        if (l == 1)
            copy_kernel<<<(nlv + 255) / 256, 256, 0, stream>>>(lv2, lv1, nlv);
        layer_kernel<<<ntiles, 128, 0, stream>>>(
            lrep, rrep, eli, eri, ediff, proj,
            packed + (size_t)l * (2 * 4 * 8 * 32 * 16),
            bin1 + l * kD, bin2 + l * kD,
            Ws1a + l * 3 * kDQ, bs1a + l * kDQ, Ws1b + l * kDQ, bs1b + l,
            Ws2a + l * 3 * kDQ, bs2a + l * kDQ, Ws2b + l * kDQ, bs2b + l,
            lvin, lvout, E);
    }
    head_kernel<<<(Nl * 3 + 3) / 4, 128, 0, stream>>>(lv2, Wl1, bl1, Wl2, bl2, lvi,
                                                      (float*)d_out, Nl * 3);
}
